// MultiHeadAttention_71914932404449
// MI455X (gfx1250) — compile-verified, matrix operand order checked by reading, not run on hardware
//
#include <hip/hip_runtime.h>

// ---------------------------------------------------------------------------
// Multi-head attention for MI455X (gfx1250, wave32, WMMA f16 16x16x32).
//   1) wcvt:        W{q,k,v,o} f32 -> f16 transposed (Wt[n][k]) in ws
//   2) gemm512<0>:  Q = X*Wq+b -> f16 [B,H,S,DK]
//      gemm512<0>:  K = X*Wk+b -> f16 [B,H,S,DK]
//      gemm512<1>:  V = X*Wv+b -> f16 [B,H,DK,S]
//   3) flash_attn:  TRANSPOSED flash attention: S^T = K*Q^T, O^T = V^T*P^T.
//                   Softmax axis is in-lane; P^T needs no layout change (the
//                   C-layout of S^T *is* the B-layout of P^T). No LDS at all.
//                   Unscaled scores (faithful to the reference's bug).
//   4) gemm512<2>:  out = attn*Wo + b -> f32 d_out
// ---------------------------------------------------------------------------

typedef __attribute__((ext_vector_type(16))) _Float16 v16h;
typedef __attribute__((ext_vector_type(8)))  _Float16 v8h;
typedef __attribute__((ext_vector_type(8)))  float    v8f;

#define BB 2
#define SS 4096
#define EE 512
#define HH 8
#define DK 64

__device__ __forceinline__ v16h cat8(v8h lo, v8h hi) {
  return __builtin_shufflevector(lo, hi, 0,1,2,3,4,5,6,7,8,9,10,11,12,13,14,15);
}

// Load one 16-row x 32-col f16 WMMA operand (A or B: same per-lane pattern):
// two contiguous 8-elem (16 B) runs at +8*half and +16+8*half.
__device__ __forceinline__ v16h ld_op(const _Float16* __restrict__ p, int half) {
  return cat8(*(const v8h*)(p + 8 * half), *(const v8h*)(p + 16 + 8 * half));
}

__device__ __forceinline__ v8h cvt8(float4 a, float4 b) {
  v8h r;
  r[0] = (_Float16)a.x; r[1] = (_Float16)a.y; r[2] = (_Float16)a.z; r[3] = (_Float16)a.w;
  r[4] = (_Float16)b.x; r[5] = (_Float16)b.y; r[6] = (_Float16)b.z; r[7] = (_Float16)b.w;
  return r;
}

// W (512x512 f32) -> WT f16 transposed: WT[n*512+k] = W[k*512+n]
__global__ void wcvt(const float* __restrict__ W, _Float16* __restrict__ WT) {
  int idx = blockIdx.x * 256 + threadIdx.x;
  int k = idx >> 9, n = idx & 511;
  WT[(size_t)n * 512 + k] = (_Float16)W[(size_t)k * 512 + n];
}

// Out[M=8192, N=512] = A[M,512] * WT^T + bias
// MODE 0: A=f32, out f16 -> [B,H,S,DK];  MODE 1: A=f32, out f16 -> [B,H,DK,S]
// MODE 2: A=f16, out f32 -> row-major [M,512] (d_out)
template <int MODE>
__global__ __launch_bounds__(128) void gemm512(const void* __restrict__ Ap,
                                               const _Float16* __restrict__ WT,
                                               const float* __restrict__ bias,
                                               void* __restrict__ Outp) {
  const int lane   = threadIdx.x & 31;
  const int wave   = threadIdx.x >> 5;
  const int lane16 = lane & 15;
  const int half   = lane >> 4;
  const int rowbase = (blockIdx.y * 4 + wave) * 16;
  const int colbase = blockIdx.x * 64;
  const int row = rowbase + lane16;

  const float*    Af = (const float*)Ap;
  const _Float16* Ah = (const _Float16*)Ap;

  v8f acc[4] = {v8f{}, v8f{}, v8f{}, v8f{}};

  for (int kc = 0; kc < 512; kc += 32) {
    v16h a;
    if (MODE == 2) {
      a = ld_op(Ah + (size_t)row * 512 + kc, half);
    } else {
      const float* ap = Af + (size_t)row * 512 + kc + 8 * half;
      float4 f0 = *(const float4*)ap,        f1 = *(const float4*)(ap + 4);
      float4 f2 = *(const float4*)(ap + 16), f3 = *(const float4*)(ap + 20);
      a = cat8(cvt8(f0, f1), cvt8(f2, f3));
    }
#pragma unroll
    for (int nt = 0; nt < 4; nt++) {
      int n = colbase + nt * 16 + lane16;
      v16h b = ld_op(WT + (size_t)n * 512 + kc, half);
      acc[nt] = __builtin_amdgcn_wmma_f32_16x16x32_f16(false, a, false, b,
                                                       (short)0, acc[nt], false, false);
    }
  }

#pragma unroll
  for (int nt = 0; nt < 4; nt++) {
    int n = colbase + nt * 16 + lane16;
    float bv = bias[n];
#pragma unroll
    for (int r = 0; r < 8; r++) {
      int   m   = rowbase + r + 8 * half;
      float val = acc[nt][r] + bv;
      if (MODE == 2) {
        ((float*)Outp)[(size_t)m * 512 + n] = val;
      } else {
        int b_ = m >> 12, s_ = m & 4095;
        int h_ = n >> 6,  d_ = n & 63;
        _Float16 hv = (_Float16)val;
        if (MODE == 0)
          ((_Float16*)Outp)[(((size_t)(b_ * HH + h_)) * SS + s_) * DK + d_] = hv;
        else
          ((_Float16*)Outp)[(((size_t)(b_ * HH + h_)) * DK + d_) * SS + s_] = hv;
      }
    }
  }
}

// Transposed flash attention. One wave = 16 query columns, 32-key steps.
// Register double-buffering on K and V tiles; no LDS; 2 cross-lane ops/step.
__global__ __launch_bounds__(128) void flash_attn(const _Float16* __restrict__ Qh,
                                                  const _Float16* __restrict__ Kh,
                                                  const _Float16* __restrict__ Vt,
                                                  _Float16* __restrict__ On) {
  const int lane   = threadIdx.x & 31;
  const int wave   = threadIdx.x >> 5;
  const int lane16 = lane & 15;
  const int half   = lane >> 4;
  const int bh     = blockIdx.x;                 // b*H + h
  const int qbase  = (blockIdx.y * 4 + wave) * 16;

  const _Float16* Kb = Kh + (size_t)bh * SS * DK;
  const _Float16* Vb = Vt + (size_t)bh * DK * SS;

  // Q as B-operand (lane = query column), dk chunks [0,32) and [32,64)
  const _Float16* qp = Qh + (size_t)bh * SS * DK + (size_t)(qbase + lane16) * DK;
  v16h bq0 = ld_op(qp, half);
  v16h bq1 = ld_op(qp + 32, half);

  v8f o[4] = {v8f{}, v8f{}, v8f{}, v8f{}};   // O^T: 4 dk-row tiles x 16 queries
  float mstat = -3.0e38f, lstat = 0.0f;      // per-lane (per-query) stats

  // ---- preload K/V tiles for kt = 0 ----------------------------------------
  // K A-operands: [t][c] = key subtile t (16 rows), dk chunk c
  v16h ak[2][2], av[4];
#pragma unroll
  for (int t = 0; t < 2; t++) {
    const _Float16* kp = Kb + (size_t)(t * 16 + lane16) * DK;
    ak[t][0] = ld_op(kp, half);
    ak[t][1] = ld_op(kp + 32, half);
  }
#pragma unroll
  for (int nt = 0; nt < 4; nt++)
    av[nt] = ld_op(Vb + (size_t)(nt * 16 + lane16) * SS, half);

  for (int kt = 0; kt < SS; kt += 32) {
    const int ktn = (kt + 32 < SS) ? kt + 32 : 0;   // clamped: branch-free pipeline

    // ---- S^T tiles: st[t] = K(16x64) * Q^T(64x16), C: row=key, lane=query ---
    v8f st0 = v8f{}, st1 = v8f{};
    st0 = __builtin_amdgcn_wmma_f32_16x16x32_f16(false, ak[0][0], false, bq0, (short)0, st0, false, false);
    st0 = __builtin_amdgcn_wmma_f32_16x16x32_f16(false, ak[0][1], false, bq1, (short)0, st0, false, false);
    st1 = __builtin_amdgcn_wmma_f32_16x16x32_f16(false, ak[1][0], false, bq0, (short)0, st1, false, false);
    st1 = __builtin_amdgcn_wmma_f32_16x16x32_f16(false, ak[1][1], false, bq1, (short)0, st1, false, false);

    // ---- issue next K loads (hidden behind softmax VALU + PV WMMAs) --------
    v16h nk[2][2];
#pragma unroll
    for (int t = 0; t < 2; t++) {
      const _Float16* kp = Kb + (size_t)(ktn + t * 16 + lane16) * DK;
      __builtin_prefetch(kp + 32 * DK, 0, 3);       // global_prefetch_b8, kt+64
      nk[t][0] = ld_op(kp, half);
      nk[t][1] = ld_op(kp + 32, half);
    }

    // ---- online softmax, fully in-lane except two xor-16 exchanges ----------
    float tm = st0[0];
#pragma unroll
    for (int r = 1; r < 8; r++) tm = fmaxf(tm, st0[r]);
#pragma unroll
    for (int r = 0; r < 8; r++) tm = fmaxf(tm, st1[r]);
    tm = fmaxf(tm, __shfl_xor(tm, 16));             // other 16 keys of this query
    const float mn   = fmaxf(mstat, tm);
    const float corr = __expf(mstat - mn);
    mstat = mn;

    // P^T B-operand: C-layout of S^T is exactly B-layout of P^T (in-lane!)
    v16h pb;
    float ps = 0.0f;
#pragma unroll
    for (int r = 0; r < 8; r++) {
      float p0 = __expf(st0[r] - mn);
      float p1 = __expf(st1[r] - mn);
      ps += p0 + p1;
      pb[r]     = (_Float16)p0;
      pb[8 + r] = (_Float16)p1;
    }
    ps += __shfl_xor(ps, 16);
    lstat = lstat * corr + ps;

    // ---- rescale + O^T += V^T * P^T ----------------------------------------
#pragma unroll
    for (int nt = 0; nt < 4; nt++) {
#pragma unroll
      for (int r = 0; r < 8; r++) o[nt][r] *= corr;
      o[nt] = __builtin_amdgcn_wmma_f32_16x16x32_f16(false, av[nt], false, pb,
                                                     (short)0, o[nt], false, false);
    }

    // ---- issue next V loads, rotate double buffers -------------------------
#pragma unroll
    for (int nt = 0; nt < 4; nt++)
      av[nt] = ld_op(Vb + (size_t)(nt * 16 + lane16) * SS + ktn, half);
#pragma unroll
    for (int t = 0; t < 2; t++) { ak[t][0] = nk[t][0]; ak[t][1] = nk[t][1]; }
  }

  // ---- normalize, store attn concat [B,S,E]: 8 consecutive dk per pack -----
  const int b_ = bh >> 3, h_ = bh & 7;
  const int s_ = qbase + lane16;                    // query index (per lane)
  const float inv = 1.0f / lstat;
  _Float16* orow = On + ((size_t)b_ * SS + s_) * EE + h_ * DK + 8 * half;
#pragma unroll
  for (int nt = 0; nt < 4; nt++) {
    v8h w;
#pragma unroll
    for (int r = 0; r < 8; r++) w[r] = (_Float16)(o[nt][r] * inv);
    *(v8h*)(orow + nt * 16) = w;                    // dk = nt*16 + 8*half + r
  }
}

extern "C" void kernel_launch(void* const* d_in, const int* in_sizes, int n_in,
                              void* d_out, int out_size, void* d_ws, size_t ws_size,
                              hipStream_t stream) {
  const float* query = (const float*)d_in[0];
  const float* keyin = (const float*)d_in[1];
  const float* value = (const float*)d_in[2];
  const float* Wq = (const float*)d_in[3];
  const float* bq = (const float*)d_in[4];
  const float* Wk = (const float*)d_in[5];
  const float* bk = (const float*)d_in[6];
  const float* Wv = (const float*)d_in[7];
  const float* bv = (const float*)d_in[8];
  const float* Wo = (const float*)d_in[9];
  const float* bo = (const float*)d_in[10];

  _Float16* WqT = (_Float16*)d_ws;
  _Float16* WkT = WqT + (size_t)512 * 512;
  _Float16* WvT = WkT + (size_t)512 * 512;
  _Float16* WoT = WvT + (size_t)512 * 512;
  _Float16* Qh  = WoT + (size_t)512 * 512;          // [B,H,S,DK]
  _Float16* Kh  = Qh + (size_t)BB * SS * EE;        // [B,H,S,DK]
  _Float16* Vt  = Kh + (size_t)BB * SS * EE;        // [B,H,DK,S]
  _Float16* An  = Vt + (size_t)BB * SS * EE;        // [B,S,E]

  wcvt<<<1024, 256, 0, stream>>>(Wq, WqT);
  wcvt<<<1024, 256, 0, stream>>>(Wk, WkT);
  wcvt<<<1024, 256, 0, stream>>>(Wv, WvT);
  wcvt<<<1024, 256, 0, stream>>>(Wo, WoT);

  dim3 gg(512 / 64, (BB * SS) / 64);
  gemm512<0><<<gg, 128, 0, stream>>>(query, WqT, bq, Qh);
  gemm512<0><<<gg, 128, 0, stream>>>(keyin, WkT, bk, Kh);
  gemm512<1><<<gg, 128, 0, stream>>>(value, WvT, bv, Vt);

  flash_attn<<<dim3(BB * HH, SS / 64), 128, 0, stream>>>(Qh, Kh, Vt, An);

  gemm512<2><<<gg, 128, 0, stream>>>(An, WoT, bo, d_out);
}
